// LogicLayer_86277303042366
// MI455X (gfx1250) — matrix operand order repaired and checked by reading, not compiled, not run on hardware
//
#include <hip/hip_runtime.h>
#include <stdint.h>

#define IN_DIM  32768
#define OUT_DIM 32768
#define BATCH   2048
#define ROWS    2      // batch rows staged in LDS per block (2 * 128KB = 256KB <= 320KB)
#define TPB     1024   // 32 waves (wave32)

typedef __attribute__((ext_vector_type(16))) _Float16 v16h;
typedef __attribute__((ext_vector_type(8)))  float    v8f;
typedef __attribute__((ext_vector_type(2)))  float    f32x2;
typedef __attribute__((__vector_size__(16))) int      i32x4;   // matches builtin param type

#define AS1 __attribute__((address_space(1)))
#define AS3 __attribute__((address_space(3)))

// GATE_COEFFS[k][n], k = gate 0..15, n = {c0, ca, cb, cab}
__device__ __constant__ float GATE[16][4] = {
  {0.f, 0.f, 0.f, 0.f}, {0.f, 0.f, 0.f, 1.f}, {0.f, 1.f, 0.f,-1.f}, {0.f, 1.f, 0.f, 0.f},
  {0.f, 0.f, 1.f,-1.f}, {0.f, 0.f, 1.f, 0.f}, {0.f, 1.f, 1.f,-2.f}, {0.f, 1.f, 1.f,-1.f},
  {1.f,-1.f,-1.f, 1.f}, {1.f,-1.f,-1.f, 2.f}, {1.f, 0.f,-1.f, 0.f}, {1.f, 0.f,-1.f, 1.f},
  {1.f,-1.f, 0.f, 0.f}, {1.f,-1.f, 0.f, 1.f}, {1.f, 0.f, 0.f,-1.f}, {1.f, 0.f, 0.f, 0.f}};

// ---------------- CDNA5 async global -> LDS copy helpers ----------------
__device__ __forceinline__ void async_copy_b128(const float* base, uint32_t byte_off,
                                                uint32_t lds_byte_addr) {
#if defined(__has_builtin) && __has_builtin(__builtin_amdgcn_global_load_async_to_lds_b128)
  __builtin_amdgcn_global_load_async_to_lds_b128(
      (AS1 i32x4*)((uintptr_t)base + byte_off),
      (AS3 i32x4*)lds_byte_addr, /*imm offset*/0, /*cpol*/0);
#else
  // VDST = LDS byte address, VADDR = 32-bit byte offset, SADDR = 64-bit base
  asm volatile("global_load_async_to_lds_b128 %0, %1, %2"
               :: "v"(lds_byte_addr), "v"(byte_off), "s"(base) : "memory");
#endif
}

__device__ __forceinline__ void wait_async0() {
#if defined(__has_builtin) && __has_builtin(__builtin_amdgcn_s_wait_asynccnt)
  __builtin_amdgcn_s_wait_asynccnt(0);
#else
  asm volatile("s_wait_asynccnt 0" ::: "memory");
#endif
}

// ---------------- Kernel 1: softmax(weights) @ GATE via WMMA + index packing ----
// 256 threads = 8 waves per block; each wave produces coeffs for 16 gate rows with
// one v_wmma_f32_16x16x32_f16 (K=0..15 live, K=16..31 zero-padded).
__global__ __launch_bounds__(256) void coeff_kernel(
    const float* __restrict__ w,       // (OUT_DIM, 16)
    const int*   __restrict__ ida,
    const int*   __restrict__ idb,
    float*       __restrict__ coeffs,  // (OUT_DIM, 4)
    int4*        __restrict__ idxp)    // OUT_DIM/2 packed (ia0,ib0,ia1,ib1)
{
  const int gid  = blockIdx.x * blockDim.x + threadIdx.x;
  const int lane = threadIdx.x & 31;
  const int wave = gid >> 5;
  const int base = wave * 16;          // this wave's 16 gate rows
  const int row  = base + (lane & 15); // A-matrix row M = lane%16
  const int half = lane >> 4;          // 0: K 0..7 ; 1: K 8..15 (ISA A-layout)
  const int col  = lane & 15;          // B/D column N

  // Load this lane's 8 of the row's 16 logits.
  const float* wr = w + (size_t)row * 16 + half * 8;
  float4 u0 = ((const float4*)wr)[0];
  float4 u1 = ((const float4*)wr)[1];
  float v[8] = {u0.x, u0.y, u0.z, u0.w, u1.x, u1.y, u1.z, u1.w};

  // Row softmax split across lane pair (L, L+16).
  float m = v[0];
#pragma unroll
  for (int i = 1; i < 8; ++i) m = fmaxf(m, v[i]);
  m = fmaxf(m, __shfl_xor(m, 16, 32));
  float e[8], s = 0.f;
#pragma unroll
  for (int i = 0; i < 8; ++i) { e[i] = __expf(v[i] - m); s += e[i]; }
  s += __shfl_xor(s, 16, 32);
  const float inv = 1.0f / s;          // TAU == 1.0

  // A (16x32 f16): lane elements 0..7 = K {half*8 .. half*8+7}; 8..15 = K 16.. (zero pad)
  v16h a;
#pragma unroll
  for (int i = 0; i < 8; ++i) a[i] = (_Float16)(e[i] * inv);
#pragma unroll
  for (int i = 8; i < 16; ++i) a[i] = (_Float16)0.f;

  // B (32x16 f16): lane column n = lane%16; elements 0..7 = K {half*8 .. half*8+7}; 8..15 = K 16.. (zero)
  v16h bm;
#pragma unroll
  for (int k = 0; k < 16; ++k) {
    float g = (k < 8 && col < 4) ? GATE[half * 8 + k][col] : 0.f;
    bm[k] = (_Float16)g;
  }

  v8f c = {};
  c = __builtin_amdgcn_wmma_f32_16x16x32_f16(false, a, false, bm, (short)0, c, false, false);

  // D layout: vgpr r, lanes 0-15 -> D[r][n]; lanes 16-31 -> D[8+r][n]
  if (col < 4) {
#pragma unroll
    for (int r = 0; r < 8; ++r) {
      int mrow = base + half * 8 + r;
      coeffs[mrow * 4 + col] = c[r];
    }
  }

  // Pack (and range-mask) index pairs: idxp[p] covers outputs 2p, 2p+1.
  if (gid < OUT_DIM / 2) {
    int4 pk;
    pk.x = ida[2 * gid]     & (IN_DIM - 1);
    pk.y = idb[2 * gid]     & (IN_DIM - 1);
    pk.z = ida[2 * gid + 1] & (IN_DIM - 1);
    pk.w = idb[2 * gid + 1] & (IN_DIM - 1);
    idxp[gid] = pk;
  }
}

// ---------------- Kernel 2: main gather/eval, LDS-staged x rows ----------------
__global__ __launch_bounds__(TPB, 1) void logic_main(
    const float*  __restrict__ x,      // (BATCH, IN_DIM)
    const int4*   __restrict__ idx,    // packed pairs
    const float4* __restrict__ cf,     // (OUT_DIM) float4 coeffs
    float*        __restrict__ out)    // (BATCH, OUT_DIM)
{
  extern __shared__ float smem[];      // ROWS * IN_DIM floats (256 KB), dynamic
  const int t  = threadIdx.x;
  const int r0 = blockIdx.x * ROWS;
  const float* src = x + (size_t)r0 * IN_DIM;
  const uint32_t lds0 = (uint32_t)(uintptr_t)&smem[0];

  // Phase 1: stage 2 contiguous rows (65536 floats) into LDS, 16 b128 chunks/thread.
#pragma unroll
  for (int k = 0; k < (ROWS * IN_DIM * 4) / (16 * TPB); ++k) {  // 16 iters
    uint32_t off = (uint32_t)(t + k * TPB) * 16u;
    async_copy_b128(src, off, lds0 + off);
  }
  wait_async0();
  __syncthreads();

  // Phase 2: each thread evaluates 2 adjacent gates for both staged rows.
  float* out0 = out + (size_t)r0 * OUT_DIM;
  float* out1 = out0 + OUT_DIM;
#pragma unroll 2
  for (int k = 0; k < (OUT_DIM / 2) / TPB; ++k) {               // 16 iters
    const int p = t + k * TPB;                                  // pair index
    const int4   ij = idx[p];
    const float4 c0 = cf[2 * p];
    const float4 c1 = cf[2 * p + 1];

    float a00 = smem[ij.x],          b00 = smem[ij.y];
    float a01 = smem[ij.z],          b01 = smem[ij.w];
    float a10 = smem[IN_DIM + ij.x], b10 = smem[IN_DIM + ij.y];
    float a11 = smem[IN_DIM + ij.z], b11 = smem[IN_DIM + ij.w];

    f32x2 o0, o1;
    o0.x = fmaf(b00, fmaf(c0.w, a00, c0.z), fmaf(c0.y, a00, c0.x));
    o0.y = fmaf(b01, fmaf(c1.w, a01, c1.z), fmaf(c1.y, a01, c1.x));
    o1.x = fmaf(b10, fmaf(c0.w, a10, c0.z), fmaf(c0.y, a10, c0.x));
    o1.y = fmaf(b11, fmaf(c1.w, a11, c1.z), fmaf(c1.y, a11, c1.x));

    __builtin_nontemporal_store(o0, (f32x2*)out0 + p);  // don't pollute L2
    __builtin_nontemporal_store(o1, (f32x2*)out1 + p);
  }
}

extern "C" void kernel_launch(void* const* d_in, const int* in_sizes, int n_in,
                              void* d_out, int out_size, void* d_ws, size_t ws_size,
                              hipStream_t stream) {
  (void)in_sizes; (void)n_in; (void)out_size; (void)ws_size;
  const float* x  = (const float*)d_in[0];
  const int*   ia = (const int*)d_in[1];
  const int*   ib = (const int*)d_in[2];
  const float* w  = (const float*)d_in[3];
  float* out = (float*)d_out;

  float* coeffs = (float*)d_ws;                                   // 512 KB
  int4*  idxp   = (int4*)((char*)d_ws + (size_t)OUT_DIM * 16);    // 256 KB

  coeff_kernel<<<dim3(OUT_DIM / 128), dim3(256), 0, stream>>>(w, ia, ib, coeffs, idxp);
  logic_main<<<dim3(BATCH / ROWS), dim3(TPB), ROWS * IN_DIM * sizeof(float), stream>>>(
      x, idxp, (const float4*)coeffs, out);
}
